// GPT_46265387712765
// MI455X (gfx1250) — matrix operand order repaired and checked by reading, not compiled, not run on hardware
//
#include <hip/hip_runtime.h>
#include <hip/hip_bf16.h>
#include <cmath>

// ---------------- types ----------------
typedef __bf16 bf16;
typedef __attribute__((ext_vector_type(16))) __bf16 v16bf;
typedef __attribute__((ext_vector_type(8)))  __bf16 v8bf;
typedef __attribute__((ext_vector_type(8)))  float  v8f;

// ---------------- model dims ----------------
#define B_   8
#define S_   512
#define E_   384
#define H_   7
#define D_   54
#define L_   7
#define V_   32000
#define FF_  2304
#define DP_  64          // padded head dim
#define HDP_ 448         // H_ * DP_
#define BS_  4096        // B_ * S_
#define BH_  56          // B_ * H_

enum { F_BF16 = 1, F_TRANS = 2, F_BIAS = 4, F_GELU = 8, F_RES = 16 };

struct Trip { int div; long so; long si; };              // off = (z/div)*so + (z%div)*si
static inline Trip T(int div, long so, long si) { Trip t; t.div = div; t.so = so; t.si = si; return t; }

struct GemmP {
  const bf16* A; const bf16* Bt; void* C; const float* bias;
  float alpha; int K, lda, ldb, ldc, flags;
  Trip at, bt, ct, bi;
};

// ---------------- bf16 WMMA GEMM: C[M,N] = A[M,K] * Bt[N,K]^T ----------------
// Block tile 256x64 (8 waves), wave tile 32x64: 2 A-frags x 4 B-frags = 8 WMMA/K-step.
#define BM 256
#define BN 64
#define BK 32
#define LDA_S 40
#define LDB_S 40

__global__ __launch_bounds__(256) void gemm_wmma_bf16(GemmP p) {
  __shared__ bf16 sA[BM * LDA_S];
  __shared__ bf16 sB[BN * LDB_S];
  const int tid  = threadIdx.x;
  const int wave = tid >> 5;
  const int lane = tid & 31;
  const int z    = blockIdx.z;

  const bf16* A  = p.A  + (long)(z / p.at.div) * p.at.so + (long)(z % p.at.div) * p.at.si
                        + (long)blockIdx.y * BM * p.lda;
  const bf16* Bt = p.Bt + (long)(z / p.bt.div) * p.bt.so + (long)(z % p.bt.div) * p.bt.si
                        + (long)blockIdx.x * BN * p.ldb;

  v8f acc[2][4];
  #pragma unroll
  for (int i = 0; i < 2; ++i)
    #pragma unroll
    for (int j = 0; j < 4; ++j)
      #pragma unroll
      for (int r = 0; r < 8; ++r) acc[i][j][r] = 0.0f;

  // A tile: 256 rows x 32 cols bf16 = 16KB; one thread = one row = 4x b128.
  // B tile:  64 rows x 32 cols bf16 =  4KB; 4 threads/row, 1x b128 each.
  const int br = tid >> 2, bq = (tid & 3) * 8;
  const int mrow0 = wave * 32 + (lane & 15);      // first A-frag row in tile
  const int kloA = (lane < 16) ? 0 : 8;           // A frag: K {0..7,16..23} | {8..15,24..31}
  const int kloB = (lane < 16) ? 0 : 8;           // B frag: K {0..7,16..23} | {8..15,24..31}

  for (int kk = 0; kk < p.K; kk += BK) {
    const bf16* as = A + (long)tid * p.lda + kk;
    *(uint4*)&sA[tid * LDA_S +  0] = *(const uint4*)(as);
    *(uint4*)&sA[tid * LDA_S +  8] = *(const uint4*)(as + 8);
    *(uint4*)&sA[tid * LDA_S + 16] = *(const uint4*)(as + 16);
    *(uint4*)&sA[tid * LDA_S + 24] = *(const uint4*)(as + 24);
    const bf16* bs = Bt + (long)br * p.ldb + kk + bq;
    *(uint4*)&sB[br * LDB_S + bq] = *(const uint4*)(bs);
    __syncthreads();

    union FR { v16bf v; v8bf h[2]; } a0, a1, b;
    a0.h[0] = *(const v8bf*)&sA[mrow0 * LDA_S + kloA];
    a0.h[1] = *(const v8bf*)&sA[mrow0 * LDA_S + kloA + 16];
    a1.h[0] = *(const v8bf*)&sA[(mrow0 + 16) * LDA_S + kloA];
    a1.h[1] = *(const v8bf*)&sA[(mrow0 + 16) * LDA_S + kloA + 16];
    #pragma unroll
    for (int j = 0; j < 4; ++j) {
      const int n = j * 16 + (lane & 15);
      b.h[0] = *(const v8bf*)&sB[n * LDB_S + kloB];
      b.h[1] = *(const v8bf*)&sB[n * LDB_S + kloB + 16];
      acc[0][j] = __builtin_amdgcn_wmma_f32_16x16x32_bf16(false, a0.v, false, b.v,
                                                          (short)0, acc[0][j], false, false);
      acc[1][j] = __builtin_amdgcn_wmma_f32_16x16x32_bf16(false, a1.v, false, b.v,
                                                          (short)0, acc[1][j], false, false);
    }
    __syncthreads();
  }

  // epilogue
  const long coff = (long)(z / p.ct.div) * p.ct.so + (long)(z % p.ct.div) * p.ct.si;
  const long boff = (long)(z / p.bi.div) * p.bi.so + (long)(z % p.bi.div) * p.bi.si;
  const int n0 = blockIdx.x * BN + (lane & 15);
  #pragma unroll
  for (int i = 0; i < 2; ++i) {
    const int m0 = blockIdx.y * BM + wave * 32 + i * 16 + ((lane >> 4) * 8);
    #pragma unroll
    for (int j = 0; j < 4; ++j) {
      const int n = n0 + j * 16;
      float bv = 0.0f;
      if (p.flags & F_BIAS) bv = p.bias[boff + n];
      #pragma unroll
      for (int r = 0; r < 8; ++r) {
        const int m = m0 + r;
        float c = acc[i][j][r] * p.alpha + bv;
        if (p.flags & F_GELU) c = 0.5f * c * (1.0f + erff(c * 0.70710678118654752f));
        const long idx = (p.flags & F_TRANS) ? (coff + (long)n * p.ldc + m)
                                             : (coff + (long)m * p.ldc + n);
        if (p.flags & F_RES)        ((float*)p.C)[idx] += c;
        else if (p.flags & F_BF16)  ((bf16*)p.C)[idx]  = (bf16)c;
        else                        ((float*)p.C)[idx] = c;
      }
    }
  }
}

// ---------------- layernorm: fp32 in -> bf16 out ----------------
__global__ __launch_bounds__(128) void ln_kernel(const float* __restrict__ x,
                                                 const float* __restrict__ g,
                                                 const float* __restrict__ b,
                                                 bf16* __restrict__ out) {
  __shared__ float red[128];
  const int row = blockIdx.x, tid = threadIdx.x;
  const float* xr = x + (long)row * E_;
  float v[3]; float s = 0.0f;
  #pragma unroll
  for (int i = 0; i < 3; ++i) { v[i] = xr[tid + 128 * i]; s += v[i]; }
  red[tid] = s; __syncthreads();
  for (int st = 64; st > 0; st >>= 1) { if (tid < st) red[tid] += red[tid + st]; __syncthreads(); }
  const float mean = red[0] / (float)E_; __syncthreads();
  float q = 0.0f;
  #pragma unroll
  for (int i = 0; i < 3; ++i) { const float d = v[i] - mean; q += d * d; }
  red[tid] = q; __syncthreads();
  for (int st = 64; st > 0; st >>= 1) { if (tid < st) red[tid] += red[tid + st]; __syncthreads(); }
  const float rstd = rsqrtf(red[0] / (float)E_ + 1e-5f);
  bf16* o = out + (long)row * E_;
  #pragma unroll
  for (int i = 0; i < 3; ++i) {
    const int c = tid + 128 * i;
    o[c] = (bf16)((v[i] - mean) * rstd * g[c] + b[c]);
  }
}

// ---------------- causal softmax: fp32 scores -> bf16 probs ----------------
__global__ __launch_bounds__(256) void softmax_kernel(const float* __restrict__ sc,
                                                      bf16* __restrict__ pr) {
  __shared__ float red[256];
  const int row = blockIdx.x, tid = threadIdx.x;
  const int qi = row % S_;
  const float* s = sc + (long)row * S_;
  bf16* p = pr + (long)row * S_;
  float mx = -3.4e38f;
  for (int i = tid; i < S_; i += 256) if (i <= qi) mx = fmaxf(mx, s[i]);
  red[tid] = mx; __syncthreads();
  for (int st = 128; st > 0; st >>= 1) { if (tid < st) red[tid] = fmaxf(red[tid], red[tid + st]); __syncthreads(); }
  mx = red[0]; __syncthreads();
  float sum = 0.0f;
  for (int i = tid; i < S_; i += 256) if (i <= qi) sum += expf(s[i] - mx);
  red[tid] = sum; __syncthreads();
  for (int st = 128; st > 0; st >>= 1) { if (tid < st) red[tid] += red[tid + st]; __syncthreads(); }
  const float inv = 1.0f / red[0];
  for (int i = tid; i < S_; i += 256)
    p[i] = (bf16)((i <= qi) ? expf(s[i] - mx) * inv : 0.0f);
}

// ---------------- embedding ----------------
__global__ void embed_kernel(const int* __restrict__ ids, const float* __restrict__ tok,
                             const float* __restrict__ pos, float* __restrict__ x) {
  const long i = (long)blockIdx.x * 256 + threadIdx.x;
  if (i >= (long)BS_ * E_) return;
  const int e = (int)(i % E_);
  const long bs = i / E_;
  const int s = (int)(bs % S_);
  x[i] = tok[(long)ids[bs] * E_ + e] + pos[(long)s * E_ + e];
}

// ---------------- weight prep (fp32 -> bf16, transpose/pad) ----------------
__global__ void prep_headw(const float* __restrict__ W, bf16* __restrict__ Wt, int total) {
  // Wt[LH][64][E]; W[LH][E][54]
  const int i = blockIdx.x * 256 + threadIdx.x; if (i >= total) return;
  const int e = i % E_; const int r = i / E_; const int d = r % DP_; const int lh = r / DP_;
  Wt[i] = (d < D_) ? (bf16)W[((long)lh * E_ + e) * D_ + d] : (bf16)0.0f;
}
__global__ void prep_headb(const float* __restrict__ b, float* __restrict__ bp, int total) {
  const int i = blockIdx.x * 256 + threadIdx.x; if (i >= total) return;
  const int d = i % DP_; const int lh = i / DP_;
  bp[i] = (d < D_) ? b[(long)lh * D_ + d] : 0.0f;
}
__global__ void prep_wc(const float* __restrict__ W, bf16* __restrict__ Wt, int total) {
  // Wt[L][E][448]: Wt[l][e][h*64+dd] = Wc[l][h*54+dd][e]
  const int i = blockIdx.x * 256 + threadIdx.x; if (i >= total) return;
  const int k = i % HDP_; const int r = i / HDP_; const int e = r % E_; const int l = r / E_;
  const int h = k / DP_, dd = k % DP_;
  Wt[i] = (dd < D_) ? (bf16)W[((long)l * (H_ * D_) + h * D_ + dd) * E_ + e] : (bf16)0.0f;
}
__global__ void prep_w1(const float* __restrict__ W, bf16* __restrict__ Wt, int total) {
  // Wt[L][FF][E]; W1[L][E][FF]
  const int i = blockIdx.x * 256 + threadIdx.x; if (i >= total) return;
  const int e = i % E_; const int r = i / E_; const int f = r % FF_; const int l = r / FF_;
  Wt[i] = (bf16)W[((long)l * E_ + e) * FF_ + f];
}
__global__ void prep_w2(const float* __restrict__ W, bf16* __restrict__ Wt, int total) {
  // Wt[L][E][FF]; W2[L][FF][E]
  const int i = blockIdx.x * 256 + threadIdx.x; if (i >= total) return;
  const int f = i % FF_; const int r = i / FF_; const int e = r % E_; const int l = r / E_;
  Wt[i] = (bf16)W[((long)l * FF_ + f) * E_ + e];
}
__global__ void prep_wout(const float* __restrict__ W, bf16* __restrict__ Wt, long total) {
  const long i = (long)blockIdx.x * 256 + threadIdx.x; if (i >= total) return;
  const int e = (int)(i % E_); const long v = i / E_;
  Wt[i] = (bf16)W[(long)e * V_ + v];
}

// ---------------- loss ----------------
__global__ __launch_bounds__(256) void rowloss_kernel(const float* __restrict__ lg,
                                                      const int* __restrict__ tgt,
                                                      float* __restrict__ out) {
  __shared__ float red[256];
  const int row = blockIdx.x, tid = threadIdx.x;
  const float* l = lg + (long)row * V_;
  float mx = -3.4e38f;
  for (int i = tid; i < V_; i += 256) mx = fmaxf(mx, l[i]);
  red[tid] = mx; __syncthreads();
  for (int st = 128; st > 0; st >>= 1) { if (tid < st) red[tid] = fmaxf(red[tid], red[tid + st]); __syncthreads(); }
  mx = red[0]; __syncthreads();
  float sum = 0.0f;
  for (int i = tid; i < V_; i += 256) sum += expf(l[i] - mx);
  red[tid] = sum; __syncthreads();
  for (int st = 128; st > 0; st >>= 1) { if (tid < st) red[tid] += red[tid + st]; __syncthreads(); }
  if (tid == 0) out[row] = logf(red[0]) + mx - l[tgt[row]];
}
__global__ __launch_bounds__(256) void finloss_kernel(const float* __restrict__ rl,
                                                      float* __restrict__ loss) {
  __shared__ float red[256];
  const int tid = threadIdx.x;
  float s = 0.0f;
  for (int i = tid; i < BS_; i += 256) s += rl[i];
  red[tid] = s; __syncthreads();
  for (int st = 128; st > 0; st >>= 1) { if (tid < st) red[tid] += red[tid + st]; __syncthreads(); }
  if (tid == 0) loss[0] = red[0] / (float)BS_;
}

// ---------------- host-side GEMM launcher ----------------
static void gemm(hipStream_t st, dim3 grid,
                 const bf16* A, const bf16* Bt, void* C, const float* bias,
                 float alpha, int K, int lda, int ldb, int ldc, int flags,
                 Trip at, Trip bt, Trip ct, Trip bi) {
  GemmP p; p.A = A; p.Bt = Bt; p.C = C; p.bias = bias; p.alpha = alpha;
  p.K = K; p.lda = lda; p.ldb = ldb; p.ldc = ldc; p.flags = flags;
  p.at = at; p.bt = bt; p.ct = ct; p.bi = bi;
  gemm_wmma_bf16<<<grid, dim3(256), 0, st>>>(p);
}

extern "C" void kernel_launch(void* const* d_in, const int* in_sizes, int n_in,
                              void* d_out, int out_size, void* d_ws, size_t ws_size,
                              hipStream_t stream) {
  (void)in_sizes; (void)n_in; (void)out_size; (void)ws_size;
  const int*   ids  = (const int*)  d_in[0];
  const int*   tgt  = (const int*)  d_in[1];
  const float* tok  = (const float*)d_in[2];
  const float* pos  = (const float*)d_in[3];
  const float* Wq   = (const float*)d_in[4];
  const float* bq   = (const float*)d_in[5];
  const float* Wk   = (const float*)d_in[6];
  const float* bk   = (const float*)d_in[7];
  const float* Wv   = (const float*)d_in[8];
  const float* bv   = (const float*)d_in[9];
  const float* Wc   = (const float*)d_in[10];
  const float* bc   = (const float*)d_in[11];
  const float* g1   = (const float*)d_in[12];
  const float* b1   = (const float*)d_in[13];
  const float* g2   = (const float*)d_in[14];
  const float* b2   = (const float*)d_in[15];
  const float* W1   = (const float*)d_in[16];
  const float* bf1  = (const float*)d_in[17];
  const float* W2   = (const float*)d_in[18];
  const float* bf2  = (const float*)d_in[19];
  const float* gf   = (const float*)d_in[20];
  const float* bfin = (const float*)d_in[21];
  const float* Wout = (const float*)d_in[22];
  const float* bout = (const float*)d_in[23];

  char* ws = (char*)d_ws;
  size_t off = 0;
  auto alloc = [&](size_t bytes) -> void* {
    void* p = ws + off; off += (bytes + 255) & ~(size_t)255; return p;
  };
  float* x   = (float*)alloc((size_t)BS_ * E_ * 4);
  bf16*  xn  = (bf16*) alloc((size_t)BS_ * E_ * 2);
  bf16*  qb  = (bf16*) alloc((size_t)BS_ * HDP_ * 2);     // [B,S,H,64]
  bf16*  kb  = (bf16*) alloc((size_t)BS_ * HDP_ * 2);     // [B,S,H,64]
  bf16*  vT  = (bf16*) alloc((size_t)BH_ * DP_ * S_ * 2); // [B,H,64,S]
  bf16*  ob  = (bf16*) alloc((size_t)BS_ * HDP_ * 2);     // [B,S,H*64]
  bf16*  hb  = (bf16*) alloc((size_t)BS_ * FF_ * 2);
  float* sc  = (float*)alloc((size_t)BH_ * S_ * S_ * 4);
  bf16*  pb  = (bf16*) alloc((size_t)BH_ * S_ * S_ * 2);
  bf16*  wqT = (bf16*) alloc((size_t)L_ * H_ * DP_ * E_ * 2);
  bf16*  wkT = (bf16*) alloc((size_t)L_ * H_ * DP_ * E_ * 2);
  bf16*  wvT = (bf16*) alloc((size_t)L_ * H_ * DP_ * E_ * 2);
  float* bqP = (float*)alloc((size_t)L_ * H_ * DP_ * 4);
  float* bkP = (float*)alloc((size_t)L_ * H_ * DP_ * 4);
  float* bvP = (float*)alloc((size_t)L_ * H_ * DP_ * 4);
  bf16*  wcT = (bf16*) alloc((size_t)L_ * E_ * HDP_ * 2);
  bf16*  w1T = (bf16*) alloc((size_t)L_ * FF_ * E_ * 2);
  bf16*  w2T = (bf16*) alloc((size_t)L_ * E_ * FF_ * 2);
  bf16*  woT = (bf16*) alloc((size_t)V_ * E_ * 2);
  float* rl  = (float*)alloc((size_t)BS_ * 4);

  float* logits = (float*)d_out;
  float* loss   = logits + (size_t)BS_ * V_;

  // ---- one-time weight conversion/transpose/pad (bf16) ----
  {
    const int t1 = L_ * H_ * DP_ * E_;
    prep_headw<<<(t1 + 255) / 256, 256, 0, stream>>>(Wq, wqT, t1);
    prep_headw<<<(t1 + 255) / 256, 256, 0, stream>>>(Wk, wkT, t1);
    prep_headw<<<(t1 + 255) / 256, 256, 0, stream>>>(Wv, wvT, t1);
    const int t2 = L_ * H_ * DP_;
    prep_headb<<<(t2 + 255) / 256, 256, 0, stream>>>(bq, bqP, t2);
    prep_headb<<<(t2 + 255) / 256, 256, 0, stream>>>(bk, bkP, t2);
    prep_headb<<<(t2 + 255) / 256, 256, 0, stream>>>(bv, bvP, t2);
    const int t3 = L_ * E_ * HDP_;
    prep_wc<<<(t3 + 255) / 256, 256, 0, stream>>>(Wc, wcT, t3);
    const int t4 = L_ * FF_ * E_;
    prep_w1<<<(t4 + 255) / 256, 256, 0, stream>>>(W1, w1T, t4);
    prep_w2<<<(t4 + 255) / 256, 256, 0, stream>>>(W2, w2T, t4);
    const long t5 = (long)V_ * E_;
    prep_wout<<<(int)((t5 + 255) / 256), 256, 0, stream>>>(Wout, woT, t5);
  }

  // ---- embedding ----
  {
    const long n = (long)BS_ * E_;
    embed_kernel<<<(int)((n + 255) / 256), 256, 0, stream>>>(ids, tok, pos, x);
  }

  const float att_scale = 1.0f / sqrtf((float)D_);
  const Trip T0 = T(1, 0, 0);

  for (int l = 0; l < L_; ++l) {
    ln_kernel<<<BS_, 128, 0, stream>>>(x, g1 + (size_t)l * E_, b1 + (size_t)l * E_, xn);

    // q = xn @ Wq + bq   (per-head N=64, z over heads)
    gemm(stream, dim3(1, 16, H_), xn, wqT + (size_t)l * H_ * DP_ * E_, qb,
         bqP + (size_t)l * H_ * DP_, 1.0f, E_, E_, E_, HDP_, F_BF16 | F_BIAS,
         T0, T(1, (long)DP_ * E_, 0), T(1, DP_, 0), T(1, DP_, 0));
    // k
    gemm(stream, dim3(1, 16, H_), xn, wkT + (size_t)l * H_ * DP_ * E_, kb,
         bkP + (size_t)l * H_ * DP_, 1.0f, E_, E_, E_, HDP_, F_BF16 | F_BIAS,
         T0, T(1, (long)DP_ * E_, 0), T(1, DP_, 0), T(1, DP_, 0));
    // v, stored transposed as vT[b,h,d,s]   (z over B*H)
    gemm(stream, dim3(1, 2, BH_), xn, wvT + (size_t)l * H_ * DP_ * E_, vT,
         bvP + (size_t)l * H_ * DP_, 1.0f, E_, E_, E_, S_, F_BF16 | F_BIAS | F_TRANS,
         T(H_, (long)S_ * E_, 0), T(H_, 0, (long)DP_ * E_), T(1, (long)DP_ * S_, 0), T(H_, 0, DP_));

    // scores = (q @ k^T) * scale   (z over B*H)
    gemm(stream, dim3(8, 2, BH_), qb, kb, sc, nullptr,
         att_scale, DP_, HDP_, HDP_, S_, 0,
         T(H_, (long)S_ * HDP_, DP_), T(H_, (long)S_ * HDP_, DP_), T(1, (long)S_ * S_, 0), T0);

    softmax_kernel<<<BH_ * S_, 256, 0, stream>>>(sc, pb);

    // o = probs @ v   -> concat heads into ob[B,S,H*64]
    gemm(stream, dim3(1, 2, BH_), pb, vT, ob, nullptr,
         1.0f, S_, S_, S_, HDP_, F_BF16,
         T(1, (long)S_ * S_, 0), T(1, (long)DP_ * S_, 0), T(H_, (long)S_ * HDP_, DP_), T0);

    // x += o @ Wc + bc
    gemm(stream, dim3(6, 16, 1), ob, wcT + (size_t)l * E_ * HDP_, x, bc + (size_t)l * E_,
         1.0f, HDP_, HDP_, HDP_, E_, F_RES | F_BIAS, T0, T0, T0, T0);

    ln_kernel<<<BS_, 128, 0, stream>>>(x, g2 + (size_t)l * E_, b2 + (size_t)l * E_, xn);

    // h = gelu(xn @ W1 + bf1)
    gemm(stream, dim3(36, 16, 1), xn, w1T + (size_t)l * FF_ * E_, hb, bf1 + (size_t)l * FF_,
         1.0f, E_, E_, E_, FF_, F_BF16 | F_BIAS | F_GELU, T0, T0, T0, T0);

    // x += h @ W2 + bf2
    gemm(stream, dim3(6, 16, 1), hb, w2T + (size_t)l * E_ * FF_, x, bf2 + (size_t)l * E_,
         1.0f, FF_, FF_, FF_, E_, F_RES | F_BIAS, T0, T0, T0, T0);
  }

  // final LN + logits + loss
  ln_kernel<<<BS_, 128, 0, stream>>>(x, gf, bfin, xn);
  gemm(stream, dim3(500, 16, 1), xn, woT, logits, bout,
       1.0f, E_, E_, E_, V_, F_BIAS, T0, T0, T0, T0);
  rowloss_kernel<<<BS_, 256, 0, stream>>>(logits, tgt, rl);
  finloss_kernel<<<1, 256, 0, stream>>>(rl, loss);
}
